// MoEST_Misar_75926431858732
// MI455X (gfx1250) — matrix operand order repaired and checked by reading, not compiled, not run on hardware
//
#include <hip/hip_runtime.h>
#include <hip/hip_bf16.h>

// ---------------------------------------------------------------------------
// CDNA5 (gfx1250, wave32) fused MoE-ST forward, bf16 data path.
// Weights pre-transposed + converted to bf16 once; activations produced in
// bf16 by GEMM epilogues. GEMM: 512-thread blocks (16 waves, 16x64 per wave),
// ping-pong LDS (BK=64), one barrier per slice, ~100 VGPRs -> no spills.
// ---------------------------------------------------------------------------

typedef __attribute__((ext_vector_type(16))) __bf16 v16bf;
typedef __attribute__((ext_vector_type(8)))  float  v8f;
typedef unsigned short ushort_t;

#define G_OUT 2000

__device__ __forceinline__ float gelu_erf(float x) {
    return 0.5f * x * (1.0f + erff(x * 0.7071067811865476f));
}
__device__ __forceinline__ float softplus_f(float x) {
    return (x > 20.0f) ? x : log1pf(expf(x));
}
// fp32 -> bf16 bits, round-to-nearest-even
__device__ __forceinline__ ushort_t f2bf(float x) {
    unsigned u = __float_as_uint(x);
    u += 0x7FFFu + ((u >> 16) & 1u);
    return (ushort_t)(u >> 16);
}

// A 16x32 bf16 frag: lane half 0 -> K {0..7,16..23}, half 1 -> K {8..15,24..31}
__device__ __forceinline__ v16bf load_fragA(const ushort_t* base, int row, int s, int hf) {
    const uint4* p = reinterpret_cast<const uint4*>(base + row * 64 + s * 32 + hf * 8);
    union { uint4 q[2]; v16bf v; } u;
    u.q[0] = p[0];
    u.q[1] = p[2];
    return u.v;
}
// B 32x16 bf16 frag: lane half 0 -> K {0..7,16..23}, half 1 -> K {8..15,24..31}
__device__ __forceinline__ v16bf load_fragB(const ushort_t* base, int col, int s, int hf) {
    const uint4* p = reinterpret_cast<const uint4*>(base + col * 64 + s * 32 + hf * 8);
    union { uint4 q[2]; v16bf v; } u;
    u.q[0] = p[0];
    u.q[1] = p[2];
    return u.v;
}

// Epilogue modes
// 0: C = v                        (fp32)
// 1: C = gelu(v)                  (fp32)
// 2: C += v ; Cb = bf16(C)        (fp32 accum + bf16 mirror)
// 3: C  = gate * v                (fp32, MoE expert 0)
// 4: C += gate * v                (fp32, MoE experts 1..3)
// 5: NB head -> mu/theta in d_out
// 6: Cb = bf16(gelu(v))           (bf16 only)
template <int EPI>
__global__ __launch_bounds__(512, 1) void gemm_wmma_bf16(
    const ushort_t* __restrict__ A, int lda,            // bf16 [M][K]
    const ushort_t* __restrict__ Bt, int K, int Ncols,  // bf16 [Ncols][K] (pre-transposed)
    const float* __restrict__ bias,
    float* __restrict__ C, ushort_t* __restrict__ Cb, int ldc,
    const float* __restrict__ gate_p, const int* __restrict__ gate_e, int expert,
    const float* __restrict__ lib,
    float* __restrict__ mu_out, float* __restrict__ th_out)
{
    // ping-pong tiles: 128 rows/cols x 64 K, bf16
    __shared__ ushort_t sA[2][128 * 64];
    __shared__ ushort_t sB[2][128 * 64];

    const int tid  = threadIdx.x;
    const int wid  = tid >> 5;
    const int lane = tid & 31;
    const int l16  = lane & 15;
    const int hf   = lane >> 4;          // lane half (0/1)
    const int wR   = wid & 7;            // 8 x 16 rows
    const int wC   = wid >> 3;           // 2 x 64 cols
    const int blockM = blockIdx.x * 128;
    const int blockN = blockIdx.y * 128;

    v8f acc[4];
#pragma unroll
    for (int c = 0; c < 4; ++c)
#pragma unroll
        for (int r = 0; r < 8; ++r) acc[c][r] = 0.0f;

    // staging: per K-slice each thread moves 2 A-quads then 2 B-quads (reused regs)
    const int sRow = tid >> 3;           // 0..63 (+i*64)
    const int sK   = (tid & 7) * 8;      // 0..56
    // B row clamp: out-of-range B columns only feed out-of-range outputs
    // (skipped in the epilogue), so clamping keeps loads in-bounds w/o zero-fill.
    int bRow[2];
#pragma unroll
    for (int i = 0; i < 2; ++i) {
        int gc = blockN + sRow + i * 64;
        bRow[i] = (gc < Ncols) ? gc : (Ncols - 1);
    }
    uint4 q[2];

    // ---- prologue: slice 0 into buffer 0 ----
#pragma unroll
    for (int i = 0; i < 2; ++i)
        q[i] = *reinterpret_cast<const uint4*>(&A[(size_t)(blockM + sRow + i * 64) * lda + sK]);
#pragma unroll
    for (int i = 0; i < 2; ++i)
        *reinterpret_cast<uint4*>(&sA[0][(sRow + i * 64) * 64 + sK]) = q[i];
#pragma unroll
    for (int i = 0; i < 2; ++i)
        q[i] = *reinterpret_cast<const uint4*>(&Bt[(size_t)bRow[i] * K + sK]);
#pragma unroll
    for (int i = 0; i < 2; ++i)
        *reinterpret_cast<uint4*>(&sB[0][(sRow + i * 64) * 64 + sK]) = q[i];
    __syncthreads();

    int k0 = 0, cur = 0;
    for (;;) {
        const int  kn   = k0 + 64;
        const bool more = kn < K;
        const int  nxt  = cur ^ 1;

        if (more) {  // A global loads; consumed (stored) after sub-step 0
#pragma unroll
            for (int i = 0; i < 2; ++i)
                q[i] = *reinterpret_cast<const uint4*>(
                    &A[(size_t)(blockM + sRow + i * 64) * lda + kn + sK]);
            if (kn + 64 < K)  // global_prefetch_b8 of slice K+128
                __builtin_prefetch(&A[(size_t)(blockM + sRow) * lda + kn + 64 + sK], 0, 3);
        }

        // ---- sub-step 0 (K 0..31 of slice) ----
        {
            const v16bf a0 = load_fragA(sA[cur], wR * 16 + l16, 0, hf);
            v16bf bf[4];
#pragma unroll
            for (int c = 0; c < 4; ++c)
                bf[c] = load_fragB(sB[cur], wC * 64 + c * 16 + l16, 0, hf);
#pragma unroll
            for (int c = 0; c < 4; ++c)
                acc[c] = __builtin_amdgcn_wmma_f32_16x16x32_bf16(
                    false, a0, false, bf[c], (short)0, acc[c], false, false);
        }

        if (more) {
            // store A quads to nxt (safe: everyone finished reading nxt at last barrier),
            // then reuse the same registers for the B global loads
#pragma unroll
            for (int i = 0; i < 2; ++i)
                *reinterpret_cast<uint4*>(&sA[nxt][(sRow + i * 64) * 64 + sK]) = q[i];
#pragma unroll
            for (int i = 0; i < 2; ++i)
                q[i] = *reinterpret_cast<const uint4*>(&Bt[(size_t)bRow[i] * K + kn + sK]);
        }

        // ---- sub-step 1 (K 32..63 of slice) ----
        {
            const v16bf a0 = load_fragA(sA[cur], wR * 16 + l16, 1, hf);
            v16bf bf[4];
#pragma unroll
            for (int c = 0; c < 4; ++c)
                bf[c] = load_fragB(sB[cur], wC * 64 + c * 16 + l16, 1, hf);
#pragma unroll
            for (int c = 0; c < 4; ++c)
                acc[c] = __builtin_amdgcn_wmma_f32_16x16x32_bf16(
                    false, a0, false, bf[c], (short)0, acc[c], false, false);
        }

        if (!more) break;
#pragma unroll
        for (int i = 0; i < 2; ++i)
            *reinterpret_cast<uint4*>(&sB[nxt][(sRow + i * 64) * 64 + sK]) = q[i];
        __syncthreads();   // one barrier per slice
        cur = nxt;
        k0 = kn;
    }

    // ---- epilogue ----
#pragma unroll
    for (int c = 0; c < 4; ++c) {
        const int col = blockN + wC * 64 + c * 16 + l16;
        if (col >= Ncols) continue;
        const float bs = bias ? bias[col] : 0.0f;
        const int row0 = blockM + wR * 16 + hf * 8;
#pragma unroll
        for (int r = 0; r < 8; ++r) {
            const int row = row0 + r;
            const float v = acc[c][r] + bs;
            if (EPI == 0) {
                C[(size_t)row * ldc + col] = v;
            } else if (EPI == 1) {
                C[(size_t)row * ldc + col] = gelu_erf(v);
            } else if (EPI == 2) {
                const float s = C[(size_t)row * ldc + col] + v;
                C[(size_t)row * ldc + col]  = s;
                Cb[(size_t)row * ldc + col] = f2bf(s);
            } else if (EPI == 3 || EPI == 4) {
                const float g = (gate_e[row] == expert) ? gate_p[row] : 0.0f;
                if (EPI == 3) C[(size_t)row * ldc + col]  = g * v;
                else          C[(size_t)row * ldc + col] += g * v;
            } else if (EPI == 5) {  // NB head
                const int gi = col >> 1;
                if (col & 1) th_out[(size_t)row * G_OUT + gi] = softplus_f(v) + 1e-6f;
                else         mu_out[(size_t)row * G_OUT + gi] = softplus_f(v) * lib[row] + 1e-6f;
            } else {                // EPI == 6
                Cb[(size_t)row * ldc + col] = f2bf(gelu_erf(v));
            }
        }
    }
}

// ---------------- conversion kernels ----------------

// fp32 [K][C] -> bf16 [C][K] (tiled transpose, all dims multiples of 32)
__global__ __launch_bounds__(256) void cvt_transpose_k(
    const float* __restrict__ in, ushort_t* __restrict__ out, int K, int C)
{
    __shared__ float tile[32][33];
    const int cb = blockIdx.x * 32, kb = blockIdx.y * 32;
    const int tx = threadIdx.x, ty = threadIdx.y;   // 32 x 8
#pragma unroll
    for (int i = 0; i < 32; i += 8)
        tile[ty + i][tx] = in[(size_t)(kb + ty + i) * C + cb + tx];
    __syncthreads();
#pragma unroll
    for (int i = 0; i < 32; i += 8)
        out[(size_t)(cb + ty + i) * K + kb + tx] = f2bf(tile[tx][ty + i]);
}

// fp32 -> bf16 elementwise (n multiple of 4)
__global__ void cvt_bf16_k(const float* __restrict__ in, ushort_t* __restrict__ out, int n)
{
    const int i = (blockIdx.x * blockDim.x + threadIdx.x) * 4;
    if (i >= n) return;
    const float4 v = *reinterpret_cast<const float4*>(&in[i]);
    uint2 o;
    o.x = (unsigned)f2bf(v.x) | ((unsigned)f2bf(v.y) << 16);
    o.y = (unsigned)f2bf(v.z) | ((unsigned)f2bf(v.w) << 16);
    *reinterpret_cast<uint2*>(&out[i]) = o;
}

// ---------------- small VALU kernels ----------------

__global__ void encode_pe_k(const float* __restrict__ pos, const float* __restrict__ fB,
                            ushort_t* __restrict__ peb, int N)
{
    const int idx = blockIdx.x * blockDim.x + threadIdx.x;
    if (idx >= N * 128) return;
    const int t = idx >> 7, j = idx & 127;
    float x = pos[t * 3 + 0] * fB[0 * 128 + j] +
              pos[t * 3 + 1] * fB[1 * 128 + j] +
              pos[t * 3 + 2] * fB[2 * 128 + j];
    x *= 6.283185307179586f;
    peb[(size_t)t * 256 + j]       = f2bf(sinf(x));
    peb[(size_t)t * 256 + 128 + j] = f2bf(cosf(x));
}

__global__ void router_k(const float* __restrict__ z, const float* __restrict__ grad,
                         const float* __restrict__ W, const float* __restrict__ b,
                         float* __restrict__ rp, int* __restrict__ re, int N)
{
    const int t = blockIdx.x * blockDim.x + threadIdx.x;
    if (t >= N) return;
    float l0 = b[0], l1 = b[1], l2 = b[2], l3 = b[3];
    const float* zr = &z[(size_t)t * 256];
    for (int d = 0; d < 256; ++d) {
        const float zv = zr[d];
        const float* w = &W[d * 4];
        l0 += zv * w[0]; l1 += zv * w[1]; l2 += zv * w[2]; l3 += zv * w[3];
    }
    const float gv = grad[t];
    l0 += gv * W[256 * 4 + 0]; l1 += gv * W[256 * 4 + 1];
    l2 += gv * W[256 * 4 + 2]; l3 += gv * W[256 * 4 + 3];
    float lm = l0; int ei = 0;                     // first-max == jax top_k tie rule
    if (l1 > lm) { lm = l1; ei = 1; }
    if (l2 > lm) { lm = l2; ei = 2; }
    if (l3 > lm) { lm = l3; ei = 3; }
    const float s = expf(l0 - lm) + expf(l1 - lm) + expf(l2 - lm) + expf(l3 - lm);
    rp[t] = 1.0f / s;                              // softmax prob of argmax
    re[t] = ei;
}

// zb = bf16(z + zmoe)   (post-MoE residual, feeds dec1 GEMM)
__global__ void add_bf16_k(const float* __restrict__ z, const float* __restrict__ zmoe,
                           ushort_t* __restrict__ zb, int n)
{
    const int i = blockIdx.x * blockDim.x + threadIdx.x;
    if (i < n) zb[i] = f2bf(z[i] + zmoe[i]);
}

__global__ void ln_gelu_k(const float* __restrict__ u, const float* __restrict__ g,
                          const float* __restrict__ b, ushort_t* __restrict__ d, int N)
{
    const int row  = blockIdx.x * 8 + (threadIdx.x >> 5);  // wave per row
    const int lane = threadIdx.x & 31;
    if (row >= N) return;
    float x[8], s = 0.0f, sq = 0.0f;
#pragma unroll
    for (int i = 0; i < 8; ++i) {
        x[i] = u[(size_t)row * 256 + lane + i * 32];
        s += x[i]; sq += x[i] * x[i];
    }
#pragma unroll
    for (int o = 16; o >= 1; o >>= 1) {
        s  += __shfl_xor(s,  o, 32);
        sq += __shfl_xor(sq, o, 32);
    }
    const float m   = s * (1.0f / 256.0f);
    const float var = sq * (1.0f / 256.0f) - m * m;
    const float inv = rsqrtf(var + 1e-5f);
#pragma unroll
    for (int i = 0; i < 8; ++i) {
        const int c = lane + i * 32;
        d[(size_t)row * 256 + c] = f2bf(gelu_erf((x[i] - m) * inv * g[c] + b[c]));
    }
}

__global__ void heads_k(const float* __restrict__ fh, const float* __restrict__ fW2,
                        const float* __restrict__ fb2,
                        const float* __restrict__ ah, const float* __restrict__ aW2,
                        const float* __restrict__ ab2,
                        float* __restrict__ outF, float* __restrict__ outA, int N)
{
    const int t = blockIdx.x * blockDim.x + threadIdx.x;
    if (t >= N) return;
    float f = fb2[0];
    for (int k = 0; k < 64; ++k) f += fh[(size_t)t * 64 + k] * fW2[k];
    outF[t] = 1.0f / (1.0f + expf(-f));
    float a[30];
#pragma unroll
    for (int o = 0; o < 30; ++o) a[o] = ab2[o];
    for (int k = 0; k < 128; ++k) {
        const float av = ah[(size_t)t * 128 + k];
        const float* w = &aW2[k * 30];
#pragma unroll
        for (int o = 0; o < 30; ++o) a[o] += av * w[o];
    }
#pragma unroll
    for (int o = 0; o < 30; ++o) outA[(size_t)t * 30 + o] = a[o];
}

// ---------------- host-side dispatch ----------------

static void run_gemm(int epi, hipStream_t s, int Nrows,
                     const ushort_t* A, int lda, const ushort_t* Bt, int K, int Ncols,
                     const float* bias, float* C, ushort_t* Cb, int ldc,
                     const float* gp = nullptr, const int* ge = nullptr, int ex = 0,
                     const float* lib = nullptr, float* mu = nullptr, float* th = nullptr)
{
    dim3 grid(Nrows / 128, (Ncols + 127) / 128), blk(512);
    switch (epi) {
    case 0: gemm_wmma_bf16<0><<<grid, blk, 0, s>>>(A, lda, Bt, K, Ncols, bias, C, Cb, ldc, gp, ge, ex, lib, mu, th); break;
    case 1: gemm_wmma_bf16<1><<<grid, blk, 0, s>>>(A, lda, Bt, K, Ncols, bias, C, Cb, ldc, gp, ge, ex, lib, mu, th); break;
    case 2: gemm_wmma_bf16<2><<<grid, blk, 0, s>>>(A, lda, Bt, K, Ncols, bias, C, Cb, ldc, gp, ge, ex, lib, mu, th); break;
    case 3: gemm_wmma_bf16<3><<<grid, blk, 0, s>>>(A, lda, Bt, K, Ncols, bias, C, Cb, ldc, gp, ge, ex, lib, mu, th); break;
    case 4: gemm_wmma_bf16<4><<<grid, blk, 0, s>>>(A, lda, Bt, K, Ncols, bias, C, Cb, ldc, gp, ge, ex, lib, mu, th); break;
    case 5: gemm_wmma_bf16<5><<<grid, blk, 0, s>>>(A, lda, Bt, K, Ncols, bias, C, Cb, ldc, gp, ge, ex, lib, mu, th); break;
    default: gemm_wmma_bf16<6><<<grid, blk, 0, s>>>(A, lda, Bt, K, Ncols, bias, C, Cb, ldc, gp, ge, ex, lib, mu, th); break;
    }
}

static void run_transpose(hipStream_t s, const float* in, ushort_t* out, int K, int C)
{
    dim3 grid(C / 32, K / 32), blk(32, 8);
    cvt_transpose_k<<<grid, blk, 0, s>>>(in, out, K, C);
}

extern "C" void kernel_launch(void* const* d_in, const int* in_sizes, int n_in,
                              void* d_out, int out_size, void* d_ws, size_t ws_size,
                              hipStream_t stream)
{
    const float* vis      = (const float*)d_in[0];
    const float* pos      = (const float*)d_in[1];
    const float* grad     = (const float*)d_in[2];
    const float* lib      = (const float*)d_in[3];
    const float* fB       = (const float*)d_in[4];
    const float* img_W    = (const float*)d_in[5];
    const float* img_b    = (const float*)d_in[6];
    const float* pos_W    = (const float*)d_in[7];
    const float* pos_b    = (const float*)d_in[8];
    const float* router_W = (const float*)d_in[9];
    const float* router_b = (const float*)d_in[10];
    const float* exp_W1   = (const float*)d_in[11];
    const float* exp_b1   = (const float*)d_in[12];
    const float* exp_W2   = (const float*)d_in[13];
    const float* exp_b2   = (const float*)d_in[14];
    const float* dec_W1   = (const float*)d_in[15];
    const float* dec_b1   = (const float*)d_in[16];
    const float* ln_g     = (const float*)d_in[17];
    const float* ln_b     = (const float*)d_in[18];
    const float* dec_W2   = (const float*)d_in[19];
    const float* dec_b2   = (const float*)d_in[20];
    const float* func_W1  = (const float*)d_in[21];
    const float* func_b1  = (const float*)d_in[22];
    const float* func_W2  = (const float*)d_in[23];
    const float* func_b2  = (const float*)d_in[24];
    const float* align_W1 = (const float*)d_in[25];
    const float* align_b1 = (const float*)d_in[26];
    const float* align_W2 = (const float*)d_in[27];
    const float* align_b2 = (const float*)d_in[28];
    (void)n_in; (void)out_size; (void)ws_size;

    const int    N  = in_sizes[0] / 1024;   // 16384
    const size_t NN = (size_t)N;

    // ---- workspace carve-out (256B aligned) ----
    char*  base = (char*)d_ws;
    size_t off  = 0;
    auto carve = [&](size_t bytes) -> char* {
        off = (off + 255) & ~(size_t)255;
        char* p = base + off;
        off += bytes;
        return p;
    };
    float*    z     = (float*)carve(NN * 256 * 4);       // fp32 z (pre-MoE); reused as u
    float*    zmoe  = (float*)carve(NN * 256 * 4);       // fp32 MoE out; reused as fh/ah
    float*    rp    = (float*)carve(NN * 4);
    int*      re    = (int*)carve(NN * 4);
    ushort_t* peb   = (ushort_t*)carve(NN * 256 * 2);    // bf16 pe; reused as db
    ushort_t* zb    = (ushort_t*)carve(NN * 256 * 2);    // bf16 z (pre-MoE then post-MoE)
    ushort_t* visb  = (ushort_t*)carve(NN * 1024 * 2);   // bf16 vis; reused as hb
    ushort_t* posWT   = (ushort_t*)carve((size_t)256 * 256 * 2);
    ushort_t* imgWT   = (ushort_t*)carve((size_t)1024 * 256 * 2);
    ushort_t* expW1T  = (ushort_t*)carve((size_t)4 * 256 * 1024 * 2);
    ushort_t* expW2T  = (ushort_t*)carve((size_t)4 * 1024 * 256 * 2);
    ushort_t* decW1T  = (ushort_t*)carve((size_t)256 * 256 * 2);
    ushort_t* decW2T  = (ushort_t*)carve((size_t)256 * 4000 * 2);
    ushort_t* funcW1T = (ushort_t*)carve((size_t)256 * 64 * 2);
    ushort_t* alignW1T= (ushort_t*)carve((size_t)256 * 128 * 2);

    float*    u    = z;                    // dec1 pre-LN (z fp32 dead after add)
    ushort_t* db   = peb;                  // post LN+GELU (pe dead after step 2)
    ushort_t* hb   = visb;                 // expert hidden (vis dead after step 3)
    float*    fh   = zmoe;                 // [N,64]  (zmoe dead after add)
    float*    ah   = zmoe + NN * 64;       // [N,128]

    float* out   = (float*)d_out;
    float* mu    = out;
    float* theta = out + NN * G_OUT;
    float* outF  = out + 2 * NN * G_OUT;
    float* outA  = outF + NN;

    // ---- 0. one-time conversions: weights -> bf16 N-major, vis -> bf16 ----
    run_transpose(stream, pos_W,    posWT,    256,  256);
    run_transpose(stream, img_W,    imgWT,    1024, 256);
    run_transpose(stream, dec_W1,   decW1T,   256,  256);
    run_transpose(stream, dec_W2,   decW2T,   256,  4000);
    run_transpose(stream, func_W1,  funcW1T,  256,  64);
    run_transpose(stream, align_W1, alignW1T, 256,  128);
    for (int e = 0; e < 4; ++e) {
        run_transpose(stream, exp_W1 + (size_t)e * 256 * 1024, expW1T + (size_t)e * 256 * 1024, 256, 1024);
        run_transpose(stream, exp_W2 + (size_t)e * 1024 * 256, expW2T + (size_t)e * 1024 * 256, 1024, 256);
    }
    cvt_bf16_k<<<(N * 1024 / 4 + 255) / 256, 256, 0, stream>>>(vis, visb, N * 1024);

    // ---- 1. Fourier positional encoding -> peb [N,256] bf16 ----
    encode_pe_k<<<(N * 128 + 255) / 256, 256, 0, stream>>>(pos, fB, peb, N);

    // ---- 2. z = gelu(pe @ pos_W + pos_b)   (fp32) ----
    run_gemm(1, stream, N, peb, 256, posWT, 256, 256, pos_b, z, nullptr, 256);

    // ---- 3. z += vis @ img_W + img_b ; zb = bf16(z) ----
    run_gemm(2, stream, N, visb, 1024, imgWT, 1024, 256, img_b, z, zb, 256);

    // ---- 4. router: top-1 expert id + prob ----
    router_k<<<(N + 255) / 256, 256, 0, stream>>>(z, grad, router_W, router_b, rp, re, N);

    // ---- 5. MoE: zmoe = sum_e gate_e * (gelu(z@W1_e+b1_e) @ W2_e + b2_e) ----
    for (int e = 0; e < 4; ++e) {
        run_gemm(6, stream, N, zb, 256, expW1T + (size_t)e * 256 * 1024, 256, 1024,
                 exp_b1 + (size_t)e * 1024, nullptr, hb, 1024);
        run_gemm(e == 0 ? 3 : 4, stream, N, hb, 1024, expW2T + (size_t)e * 1024 * 256, 1024, 256,
                 exp_b2 + (size_t)e * 256, zmoe, nullptr, 256, rp, re, e);
    }

    // ---- 6. zb = bf16(z + zmoe) ----
    add_bf16_k<<<(N * 256 + 255) / 256, 256, 0, stream>>>(z, zmoe, zb, N * 256);

    // ---- 7. u = z_new @ dec_W1 + dec_b1   (fp32, into z's storage) ----
    run_gemm(0, stream, N, zb, 256, decW1T, 256, 256, dec_b1, u, nullptr, 256);

    // ---- 8. db = bf16(gelu(layernorm(u))) ----
    ln_gelu_k<<<N / 8, 256, 0, stream>>>(u, ln_g, ln_b, db, N);

    // ---- 9. NB head: mu/theta straight from d @ dec_W2 + dec_b2 ----
    run_gemm(5, stream, N, db, 256, decW2T, 256, 4000, dec_b2,
             nullptr, nullptr, 0, nullptr, nullptr, 0, lib, mu, theta);

    // ---- 10/11. small head hiddens (fp32 for VALU finishers) ----
    run_gemm(1, stream, N, db, 256, funcW1T, 256, 64, func_b1, fh, nullptr, 64);
    run_gemm(1, stream, N, db, 256, alignW1T, 256, 128, align_b1, ah, nullptr, 128);

    // ---- 12. final tiny projections ----
    heads_k<<<(N + 255) / 256, 256, 0, stream>>>(fh, func_W2, func_b2,
                                                 ah, align_W2, align_b2, outF, outA, N);
}
